// KAN_Convolution_74414603370735
// MI455X (gfx1250) — compile-verified, matrix operand order checked by reading, not run on hardware
//
#include <hip/hip_runtime.h>
#include <hip/hip_bf16.h>
#include <math.h>

// CDNA5 WMMA / descriptor types
typedef __attribute__((ext_vector_type(16))) _Float16 v16h;
typedef __attribute__((ext_vector_type(8)))  float    v8f;
typedef __attribute__((ext_vector_type(4)))  unsigned int v4u;
typedef __attribute__((ext_vector_type(8)))  unsigned int v8u;

#define TILE    32          // output tile per block
#define IN_T    34          // input tile (TILE + K - 1)
#define NE      (IN_T*IN_T) // 1156 input elements per tile
#define NCHUNK  80          // ceil(1156/16)=73, padded to 8-wave multiple
#define GSTRIDE 9           // floats per element in G (9 positions)
#define HH      256
#define WW      256
#define HO      254
#define WO      254

// ---------------------------------------------------------------------------
// Prep kernel: fold spline_weight (9x8) + base_weight (9) into the WMMA B
// matrix (K=32 x N=16, f16), pre-swizzled into the per-lane register layout
// for v_wmma_f32_16x16x32_f16:
//   lane<16 : N=lane,    halves 0..7 = K0..7,  halves 8..15 = K16..23
//   lane>=16: N=lane-16, halves 0..7 = K8..15, halves 8..15 = K24..31
// Feature k = 4*cell + deg (k<12) -> coeff of t^deg for that cell,
// k = 12 -> base weight (silu feature). k >= 13 -> 0.
// ---------------------------------------------------------------------------
__global__ void kan_prep_B(const float* __restrict__ base_w,
                           const float* __restrict__ spline_w,
                           _Float16* __restrict__ Bws) {
    int lane = threadIdx.x;           // 0..31
    const float M[4][4] = {
        { 1.f/6.f, -3.f/6.f,  3.f/6.f, -1.f/6.f },   // N0 = (1-t)^3/6
        { 4.f/6.f,  0.f,      -6.f/6.f,  3.f/6.f },  // N1
        { 1.f/6.f,  3.f/6.f,  3.f/6.f, -3.f/6.f },   // N2
        { 0.f,      0.f,      0.f,      1.f/6.f }    // N3 = t^3/6
    };
    int n  = lane & 15;               // output column = patch position p
    int hi = lane >> 4;               // which K-half this lane supplies
    #pragma unroll
    for (int h = 0; h < 16; ++h) {
        int k = (h < 8) ? (hi * 8 + h) : (16 + hi * 8 + (h - 8));
        float v = 0.f;
        if (n < 9 && k < 13) {
            if (k == 12) {
                v = base_w[n];
            } else {
                int c = k >> 2, d = k & 3;
                // active bases for cell c are global bases c+2 .. c+5
                #pragma unroll
                for (int j = 0; j < 4; ++j)
                    v += spline_w[n * 8 + (c + 2 + j)] * M[j][d];
            }
        }
        Bws[lane * 16 + h] = (_Float16)v;
    }
}

// ---------------------------------------------------------------------------
// Main kernel: per block, one 32x32 output tile of one batch image.
// Phase 0: TDM tensor_load_to_lds DMAs the 34x34 f32 input tile into LDS.
// Phase 1: 8 waves x 10 iters, each iter one v_wmma_f32_16x16x32_f16 turning
//          16 input elements' nonlinear features into 9 per-position
//          contributions G[elem][p], scattered to LDS.
// Phase 2: shift-add gather of the 3x3 footprint per output pixel.
// Tile origins are clamped so every DMA window / output store is in-bounds;
// overlapping edge tiles rewrite identical values (deterministic).
// ---------------------------------------------------------------------------
__launch_bounds__(256)
__global__ void kan_conv_wmma(const float* __restrict__ x,
                              const _Float16* __restrict__ Bws,
                              float* __restrict__ out) {
    __shared__ float xs[NE];                     // 4624 B input tile
    __shared__ float G[NCHUNK * 16 * GSTRIDE];   // 46080 B

    const int tw = blockIdx.x, th = blockIdx.y, b = blockIdx.z;
    int h0 = th * TILE; h0 = (h0 > HO - TILE) ? (HO - TILE) : h0;  // <=222
    int w0 = tw * TILE; w0 = (w0 > WO - TILE) ? (WO - TILE) : w0;
    const int tid  = threadIdx.x;
    const int lane = tid & 31;
    const int wave = tid >> 5;                   // 8 waves

    // Per-lane row of the pre-swizzled B matrix (uniform across waves).
    const v16h Bm = *(const v16h*)(Bws + lane * 16);

    // ---------------- Phase 0: TDM async tile load ----------------
    if (wave == 0) {
        const float* gsrc = x + (size_t)b * HH * WW + h0 * WW + w0;
        unsigned long long ga = (unsigned long long)(size_t)gsrc;
        unsigned lds_off = (unsigned)(size_t)(void*)xs;

        v4u g0;
        g0[0] = 1u;                               // count=1, no gather
        g0[1] = lds_off;                          // D#.lds_addr
        g0[2] = (unsigned)ga;                     // global_addr[31:0]
        g0[3] = (unsigned)(ga >> 32) | (2u << 30);// global_addr[56:32] | type=2

        v8u g1;
        g1[0] = 2u << 16;                         // data_size=4B; mask=0
        g1[1] = ((unsigned)WW & 0xffffu) << 16;   // tensor_dim0[15:0]=256
        g1[2] = ((unsigned)WW & 0xffffu) << 16;   // dim0[31:16]=0 | tensor_dim1[15:0]=256
        g1[3] = (unsigned)IN_T << 16;             // dim1[31:16]=0 | tile_dim0=34
        g1[4] = (unsigned)IN_T;                   // tile_dim1=34 | tile_dim2=0
        g1[5] = (unsigned)WW;                     // tensor_dim0_stride=256
        g1[6] = 0u;                               // stride0[47:32]=0 | stride1[15:0]=0
        g1[7] = (unsigned)((HH * WW) >> 16);      // tensor_dim1_stride[47:16] (65536)

        v4u g2 = {};                              // dims 2/3 unused (tile_dim2=0)
        v4u g3 = {};

        asm volatile("tensor_load_to_lds %0, %1, %2, %3"
                     :: "s"(g0), "s"(g1), "s"(g2), "s"(g3)
                     : "memory");
        __builtin_amdgcn_s_wait_tensorcnt(0);
    }
    __syncthreads();

    // ---------------- Phase 1: features + WMMA ----------------
    for (int it = 0; it < 10; ++it) {
        const int chunk = it * 8 + wave;
        int e = chunk * 16 + (lane & 15);
        int ec = (e < NE) ? e : (NE - 1);        // clamp padded lanes
        const float xv = xs[ec];

        // cell/local-coordinate for uniform cubic B-spline, x in [0,1):
        // u = (x+0.2)/0.4 in [0.5,3.0) -> cell = floor(u) in {0,1,2}, t = u-cell
        float u = (xv + 0.2f) * 2.5f;
        int cell = (int)u;
        cell = (cell < 0) ? 0 : ((cell > 2) ? 2 : cell);
        const float t  = u - (float)cell;
        const float t2 = t * t;
        const float t3 = t2 * t;
        const float sl = xv / (1.f + __expf(-xv));   // silu

        // Build A (16x32 f16) per the ISA layout.
        v16h a = {};
        if (lane < 16) {
            if (cell == 0) {
                a[0] = (_Float16)1.f; a[1] = (_Float16)t;
                a[2] = (_Float16)t2;  a[3] = (_Float16)t3;
            } else if (cell == 1) {
                a[4] = (_Float16)1.f; a[5] = (_Float16)t;
                a[6] = (_Float16)t2;  a[7] = (_Float16)t3;
            }
        } else {
            if (cell == 2) {
                a[0] = (_Float16)1.f; a[1] = (_Float16)t;
                a[2] = (_Float16)t2;  a[3] = (_Float16)t3;
            }
            a[4] = (_Float16)sl;                     // K=12: silu feature
        }

        // Uniform control flow here: EXEC is all-ones for the WMMA.
        v8f cz = {};
        v8f d = __builtin_amdgcn_wmma_f32_16x16x32_f16(
            /*neg_a=*/false, a, /*neg_b=*/false, Bm,
            /*c_mod=*/(short)0, cz, /*reuse_a=*/false, /*reuse_b=*/false);

        // Scatter D (16 elems x 9 positions) to LDS.
        // Lane holds column N=lane&15, rows M = r (+8 for high half).
        const int n = lane & 15;
        if (n < 9) {
            const int ebase = chunk * 16 + ((lane >> 4) << 3);
            #pragma unroll
            for (int r = 0; r < 8; ++r)
                G[(ebase + r) * GSTRIDE + n] = d[r];
        }
    }

    __syncthreads();

    // ---------------- Phase 2: shift-add gather ----------------
    for (int i = tid; i < TILE * TILE; i += 256) {
        const int oh = i >> 5, ow = i & 31;
        float s = 0.f;
        #pragma unroll
        for (int ka = 0; ka < 3; ++ka)
            #pragma unroll
            for (int kc = 0; kc < 3; ++kc) {
                const int e = (oh + ka) * IN_T + (ow + kc);
                s += G[e * GSTRIDE + (ka * 3 + kc)];
            }
        out[(size_t)b * HO * WO + (h0 + oh) * WO + (w0 + ow)] = s;
    }
}

// ---------------------------------------------------------------------------
extern "C" void kernel_launch(void* const* d_in, const int* in_sizes, int n_in,
                              void* d_out, int out_size, void* d_ws, size_t ws_size,
                              hipStream_t stream) {
    const float* x  = (const float*)d_in[0];   // (32,256,256)
    const float* bw = (const float*)d_in[1];   // (3,3)
    const float* sw = (const float*)d_in[2];   // (3,3,8)
    float* out = (float*)d_out;                // (32,254,254)
    _Float16* Bws = (_Float16*)d_ws;           // 32 lanes * 16 halves = 1 KB

    kan_prep_B<<<1, 32, 0, stream>>>(bw, sw, Bws);

    dim3 grid((WO + TILE - 1) / TILE, (HO + TILE - 1) / TILE, 32);  // 8,8,32
    kan_conv_wmma<<<grid, 256, 0, stream>>>(x, Bws, out);
}
